// Attention_18511309046077
// MI455X (gfx1250) — matrix operand order repaired and checked by reading, not compiled, not run on hardware
//
#include <hip/hip_runtime.h>

// ---------------------------------------------------------------------------
// Causal MHA for MI455X (gfx1250, wave32, WMMA).
// B=2, N=2048, C=768, H=12, D=64.  bf16 matmuls (v_wmma_f32_16x16x32_bf16),
// f32 accumulation + f32 softmax statistics.  Flash-attention style: no NxN
// score matrix is ever materialized in global memory.
// GEMMs are register-blocked 16x64 per wave: each A-fragment feeds 4 WMMAs.
// ---------------------------------------------------------------------------

#define BATCH 2
#define SEQ   2048
#define CDIM  768
#define NH    12
#define HD    64
#define ROWS  (BATCH * SEQ)   // 4096
#define SCALEF 0.125f         // 64^-0.5
#define FWAVES 4              // waves per flash block

typedef __attribute__((ext_vector_type(16))) __bf16 v16bf;
typedef __attribute__((ext_vector_type(8)))  __bf16 v8bf;
typedef __attribute__((ext_vector_type(8)))  float  v8f;

__device__ __forceinline__ unsigned short f2bf(float f) {
    unsigned u = __float_as_uint(f);
    u += 0x7FFFu + ((u >> 16) & 1u);          // round-to-nearest-even
    return (unsigned short)(u >> 16);
}

// A-fragment (16x32 bf16): lane<16 holds row=lane, K 0..15 contiguous;
// lane>=16 holds row=lane-16, K 16..31.  v16bf elem e -> VGPR e/2.
__device__ __forceinline__ v16bf load_a(const unsigned short* base, int ld,
                                        int row, int k0, int lane) {
    const unsigned short* p = base + (size_t)row * ld + k0 + ((lane & 16) ? 16 : 0);
    v8bf lo = *(const v8bf*)p;          // K + {0..7}
    v8bf hi = *(const v8bf*)(p + 8);    // K + {8..15}
    v16bf r;
#pragma unroll
    for (int i = 0; i < 8; ++i) { r[i] = lo[i]; r[i + 8] = hi[i]; }
    return r;
}

// B-fragment (32x16 bf16): lane<16 holds col=lane, K 0..15 contiguous;
// lane>=16 holds col=lane-16, K 16..31.  Memory is [col][K] row-major,
// which is exactly row-major W for y = x*W^T (B(k,n) = W[n][k]).
__device__ __forceinline__ v16bf load_b(const unsigned short* base, int ld,
                                        int col0, int k0, int lane) {
    const unsigned short* p =
        base + (size_t)(col0 + (lane & 15)) * ld + k0 + ((lane & 16) ? 16 : 0);
    return *(const v16bf*)p;
}

__device__ __forceinline__ v8f wmma_bf16(v16bf a, v16bf b, v8f c) {
    return __builtin_amdgcn_wmma_f32_16x16x32_bf16(
        false, a, false, b, (short)0, c, false, false);
}

// ---------------------------------------------------------------------------
__global__ void cvt_bf16(const float* __restrict__ src,
                         unsigned short* __restrict__ dst, int n) {
    int i = blockIdx.x * blockDim.x + threadIdx.x;
    int stride = gridDim.x * blockDim.x;
    for (; i < n; i += stride) dst[i] = f2bf(src[i]);
}

// ---------------------------------------------------------------------------
// Q/K/V projections.  One wave per 16x64 tile (A reused across 4 WMMAs).
// With CDIM/64 == NH, each wave's 64 columns are exactly one head (h = nt).
// Q scaled by D^-0.5, stored [b][h][n][d]; K stored [b][h][n][d];
// V stored transposed [b][h][d][n] so the PV B-fragment is contiguous.
__global__ __launch_bounds__(32) void gemm_qkv(
    const unsigned short* __restrict__ xb,
    const unsigned short* __restrict__ wq,
    const unsigned short* __restrict__ wk,
    const unsigned short* __restrict__ wv,
    unsigned short* __restrict__ qbuf,
    unsigned short* __restrict__ kbuf,
    unsigned short* __restrict__ vtbuf) {
    int lane = threadIdx.x;
    int nt = blockIdx.x;             // 0..11  (== head index)
    int mt = blockIdx.y;             // 0..255
    int which = blockIdx.z;          // 0=Q 1=K 2=V
    const unsigned short* w = (which == 0) ? wq : (which == 1) ? wk : wv;

    v8f acc[4] = {{}, {}, {}, {}};
    for (int k0 = 0; k0 < CDIM; k0 += 32) {
        v16bf a = load_a(xb, CDIM, mt * 16 + (lane & 15), k0, lane);
#pragma unroll
        for (int sub = 0; sub < 4; ++sub) {
            v16bf b = load_b(w, CDIM, nt * 64 + sub * 16, k0, lane);
            acc[sub] = wmma_bf16(a, b, acc[sub]);
        }
    }

    int h = nt;
    int col15 = lane & 15;
    int half8 = (lane & 16) ? 8 : 0;
#pragma unroll
    for (int r = 0; r < 8; ++r) {
        int row = mt * 16 + r + half8;           // global token row
        int b_ = row >> 11, n = row & (SEQ - 1);
        int bh = b_ * NH + h;
#pragma unroll
        for (int sub = 0; sub < 4; ++sub) {
            int d = sub * 16 + col15;
            float v = acc[sub][r];
            if (which == 0)
                qbuf[((size_t)bh * SEQ + n) * HD + d] = f2bf(v * SCALEF);
            else if (which == 1)
                kbuf[((size_t)bh * SEQ + n) * HD + d] = f2bf(v);
            else
                vtbuf[((size_t)bh * HD + d) * SEQ + n] = f2bf(v);
        }
    }
}

// ---------------------------------------------------------------------------
// Flash attention.  One wave per 16-row query tile; 32-key blocks.
// S(16x32) = 4 WMMAs, online softmax in f32 (half-wave shfl reductions),
// P staged in per-wave LDS (C-layout -> A-layout transpose), O += P*V = 4 WMMAs.
__global__ __launch_bounds__(FWAVES * 32) void flash_attn(
    const unsigned short* __restrict__ qbuf,
    const unsigned short* __restrict__ kbuf,
    const unsigned short* __restrict__ vtbuf,
    unsigned short* __restrict__ obuf) {
    __shared__ __align__(16) unsigned short plds[FWAVES][16][32];

    int tid = threadIdx.x;
    int lane = tid & 31;
    int wid = tid >> 5;
    int bh = blockIdx.y;                       // 0..23
    int qt = blockIdx.x * FWAVES + wid;        // 0..127
    int q0 = qt * 16;

    const unsigned short* Qh = qbuf + (size_t)bh * SEQ * HD;
    const unsigned short* Kh = kbuf + (size_t)bh * SEQ * HD;
    const unsigned short* Vh = vtbuf + (size_t)bh * HD * SEQ;

    // Q tile A-fragments (D=64 -> two k-steps), resident for the whole loop.
    v16bf qf0 = load_a(Qh, HD, q0 + (lane & 15), 0, lane);
    v16bf qf1 = load_a(Qh, HD, q0 + (lane & 15), 32, lane);

    v8f o0 = {}, o1 = {}, o2 = {}, o3 = {};
    float mrow[8], lrow[8];
#pragma unroll
    for (int r = 0; r < 8; ++r) { mrow[r] = -1e30f; lrow[r] = 0.f; }

    int half8 = (lane & 16) ? 8 : 0;
    int col = lane & 15;
    int jend = q0 + 16;                        // causal: keys <= q0+15 needed

    for (int jb = 0; jb < jend; jb += 32) {
        __builtin_prefetch(Kh + (size_t)(jb + 32) * HD, 0, 0);

        // S tile: two 16x16 column halves of the 32-key block.
        v8f s0 = {}, s1 = {};
        {
            v16bf bk;
            bk = load_b(Kh, HD, jb, 0, lane);       s0 = wmma_bf16(qf0, bk, s0);
            bk = load_b(Kh, HD, jb, 32, lane);      s0 = wmma_bf16(qf1, bk, s0);
            bk = load_b(Kh, HD, jb + 16, 0, lane);  s1 = wmma_bf16(qf0, bk, s1);
            bk = load_b(Kh, HD, jb + 16, 32, lane); s1 = wmma_bf16(qf1, bk, s1);
        }

        // Online softmax.  C-layout: lane holds col=lane&15, rows r + half8.
        // Row stats reduce across 16 lanes (xor masks 1..8 stay within half).
#pragma unroll
        for (int r = 0; r < 8; ++r) {
            int qrow = q0 + r + half8;
            float e0 = ((jb + col) > qrow) ? -1e30f : s0[r];
            float e1 = ((jb + 16 + col) > qrow) ? -1e30f : s1[r];
            float mloc = fmaxf(e0, e1);
#pragma unroll
            for (int msk = 1; msk < 16; msk <<= 1)
                mloc = fmaxf(mloc, __shfl_xor(mloc, msk, 32));
            float mn = fmaxf(mrow[r], mloc);
            float al = __expf(mrow[r] - mn);
            float p0 = __expf(e0 - mn);
            float p1 = __expf(e1 - mn);
            float ps = p0 + p1;
#pragma unroll
            for (int msk = 1; msk < 16; msk <<= 1)
                ps += __shfl_xor(ps, msk, 32);
            lrow[r] = lrow[r] * al + ps;
            mrow[r] = mn;
            o0[r] *= al; o1[r] *= al; o2[r] *= al; o3[r] *= al;
            plds[wid][r + half8][col] = f2bf(p0);
            plds[wid][r + half8][16 + col] = f2bf(p1);
        }
        // DS ops are in-order within a wave; the wait + memory clobber stops
        // the compiler from hoisting the A-fragment reads above the stores.
        asm volatile("s_wait_dscnt 0" ::: "memory");

        // P (16x32) as a single A-fragment straight from LDS.
        v16bf pf = load_a(&plds[wid][0][0], 32, (lane & 15), 0, lane);

        // O += P * V  (V transposed: [d][n], contiguous B-fragments).
        v16bf bv;
        bv = load_b(Vh, SEQ, 0, jb, lane);  o0 = wmma_bf16(pf, bv, o0);
        bv = load_b(Vh, SEQ, 16, jb, lane); o1 = wmma_bf16(pf, bv, o1);
        bv = load_b(Vh, SEQ, 32, jb, lane); o2 = wmma_bf16(pf, bv, o2);
        bv = load_b(Vh, SEQ, 48, jb, lane); o3 = wmma_bf16(pf, bv, o3);
    }

    // Epilogue: normalize and merge heads -> [b][n][C] bf16.
    int b_ = bh / NH, h = bh % NH;
#pragma unroll
    for (int r = 0; r < 8; ++r) {
        float inv = 1.f / lrow[r];
        int n = q0 + r + half8;
        size_t off = ((size_t)b_ * SEQ + n) * CDIM + h * HD;
        obuf[off + 0 + col]  = f2bf(o0[r] * inv);
        obuf[off + 16 + col] = f2bf(o1[r] * inv);
        obuf[off + 32 + col] = f2bf(o2[r] * inv);
        obuf[off + 48 + col] = f2bf(o3[r] * inv);
    }
}

// ---------------------------------------------------------------------------
// out = O @ Wo^T + bo  (f32 output).  16x64 register-blocked per wave.
__global__ __launch_bounds__(32) void gemm_out(
    const unsigned short* __restrict__ ob,
    const unsigned short* __restrict__ wo,
    const float* __restrict__ bo,
    float* __restrict__ out) {
    int lane = threadIdx.x;
    int nt = blockIdx.x;             // 0..11
    int mt = blockIdx.y;             // 0..255

    v8f acc[4] = {{}, {}, {}, {}};
    for (int k0 = 0; k0 < CDIM; k0 += 32) {
        v16bf a = load_a(ob, CDIM, mt * 16 + (lane & 15), k0, lane);
#pragma unroll
        for (int sub = 0; sub < 4; ++sub) {
            v16bf b = load_b(wo, CDIM, nt * 64 + sub * 16, k0, lane);
            acc[sub] = wmma_bf16(a, b, acc[sub]);
        }
    }

    int col15 = lane & 15;
    int half8 = (lane & 16) ? 8 : 0;
#pragma unroll
    for (int sub = 0; sub < 4; ++sub) {
        int col = nt * 64 + sub * 16 + col15;
        float bias = bo[col];
#pragma unroll
        for (int r = 0; r < 8; ++r) {
            int row = mt * 16 + r + half8;
            out[(size_t)row * CDIM + col] = acc[sub][r] + bias;
        }
    }
}

// ---------------------------------------------------------------------------
extern "C" void kernel_launch(void* const* d_in, const int* in_sizes, int n_in,
                              void* d_out, int out_size, void* d_ws, size_t ws_size,
                              hipStream_t stream) {
    const float* x  = (const float*)d_in[0];
    const float* Wq = (const float*)d_in[1];
    const float* Wk = (const float*)d_in[2];
    const float* Wv = (const float*)d_in[3];
    const float* Wo = (const float*)d_in[4];
    const float* bo = (const float*)d_in[5];
    float* out = (float*)d_out;

    char* p = (char*)d_ws;
    auto take = [&](size_t elts) -> unsigned short* {
        unsigned short* r = (unsigned short*)p;
        p += ((elts * 2 + 255) / 256) * 256;
        return r;
    };
    unsigned short* xb  = take((size_t)ROWS * CDIM);
    unsigned short* wqb = take((size_t)CDIM * CDIM);
    unsigned short* wkb = take((size_t)CDIM * CDIM);
    unsigned short* wvb = take((size_t)CDIM * CDIM);
    unsigned short* wob = take((size_t)CDIM * CDIM);
    unsigned short* qb  = take((size_t)ROWS * CDIM);   // [b][h][n][d]
    unsigned short* kb  = take((size_t)ROWS * CDIM);   // [b][h][n][d]
    unsigned short* vt  = take((size_t)ROWS * CDIM);   // [b][h][d][n]
    unsigned short* ob  = take((size_t)ROWS * CDIM);   // [b][n][C]

    cvt_bf16<<<1024, 256, 0, stream>>>(x, xb, ROWS * CDIM);
    cvt_bf16<<<512, 256, 0, stream>>>(Wq, wqb, CDIM * CDIM);
    cvt_bf16<<<512, 256, 0, stream>>>(Wk, wkb, CDIM * CDIM);
    cvt_bf16<<<512, 256, 0, stream>>>(Wv, wvb, CDIM * CDIM);
    cvt_bf16<<<512, 256, 0, stream>>>(Wo, wob, CDIM * CDIM);

    gemm_qkv<<<dim3(CDIM / 64, ROWS / 16, 3), 32, 0, stream>>>(
        xb, wqb, wkb, wvb, qb, kb, vt);

    flash_attn<<<dim3((SEQ / 16) / FWAVES, BATCH * NH), FWAVES * 32, 0, stream>>>(
        qb, kb, vt, ob);

    gemm_out<<<dim3(CDIM / 64, ROWS / 16), 32, 0, stream>>>(ob, wob, bo, out);
}
